// MultiHeadAttention_71768903516414
// MI455X (gfx1250) — matrix operand order repaired and checked by reading, not compiled, not run on hardware
//
#include <hip/hip_runtime.h>
#include <hip/hip_bf16.h>
#include <stdint.h>

// ---------------------------------------------------------------------------
// Types / WMMA helpers (CDNA5 gfx1250, wave32)
// ---------------------------------------------------------------------------
typedef __attribute__((ext_vector_type(16))) __bf16 v16bf;
typedef __attribute__((ext_vector_type(8)))  __bf16 v8bf;
typedef __attribute__((ext_vector_type(8)))  float  v8f;
typedef __attribute__((ext_vector_type(4)))  unsigned int u32x4;
typedef __attribute__((ext_vector_type(4)))  int    i32x4;
typedef __attribute__((ext_vector_type(8)))  int    i32x8;

#define WMMA_BF16(a, b, c) \
  __builtin_amdgcn_wmma_f32_16x16x32_bf16(false, (a), false, (b), (short)0, (c), false, false)

static constexpr int Bb = 4;      // batch
static constexpr int S  = 2048;   // sequence
static constexpr int D  = 1024;   // model dim
static constexpr int H  = 16;     // heads
static constexpr int DK = 64;     // head dim
static constexpr int N3 = 3 * D;  // qkv output dim

__device__ __forceinline__ __bf16 f2bf(float f) {
  union { float f; uint32_t u; } v; v.f = f;
  uint32_t r = v.u + 0x7FFFu + ((v.u >> 16) & 1u);   // round-to-nearest-even
  unsigned short hs = (unsigned short)(r >> 16);
  return __builtin_bit_cast(__bf16, hs);
}

union AFrag { v16bf v; v8bf h[2]; };

// A fragment (16x32, bf16): rows [row0,row0+16), contraction [kbase,kbase+32).
// Order used here (same as B): lane L -> row m=L&15; element e -> k = 16*hi + e  (hi = L>>4)
__device__ __forceinline__ v16bf load_a_frag(const __bf16* __restrict__ src,
                                             int row0, int ld, int kbase, int lane) {
  int m = lane & 15, hi = lane >> 4;
  const __bf16* p = src + (size_t)(row0 + m) * ld + kbase;
  AFrag f;
  f.h[0] = *(const v8bf*)(p + hi * 16);
  f.h[1] = *(const v8bf*)(p + hi * 16 + 8);
  return f.v;
}

// B fragment (32x16, bf16): B[k][n] = W[ncol0+n][kbase+k], W row-major (ld).
// Order used here (same as A): lane L -> col n=L&15; element e -> k = 16*hi + e  (contiguous 32B)
__device__ __forceinline__ v16bf load_b_frag(const __bf16* __restrict__ W,
                                             int ncol0, int ld, int kbase, int lane) {
  int n = lane & 15, hi = lane >> 4;
  const __bf16* p = W + (size_t)(ncol0 + n) * ld + kbase + hi * 16;
  return *(const v16bf*)p;
}

// ---------------------------------------------------------------------------
// Tensor Data Mover: async 2D tile (bf16) Global -> LDS.
// D# per CDNA5 ISA:
//   group0: count=1 | lds_addr | global_addr[56:0] | type=2
//   group1: data_size=1(2B), tile_dim0 (contiguous elems), tile_dim1 (rows),
//           tensor_dim0_stride = row stride (elems); tensor dims set huge so
//           no OOB clipping ever triggers. Groups 2/3 zero (2D tile).
// ---------------------------------------------------------------------------
__device__ __forceinline__ void tdm_load_2d_to_lds(const void* gsrc,
                                                   uint32_t lds_addr,
                                                   uint32_t tile_d0_elems,
                                                   uint32_t tile_d1_rows,
                                                   uint32_t row_stride_elems) {
  uint64_t ga = (uint64_t)(uintptr_t)gsrc;
  u32x4 g0 = { 1u,                                   // count=1, user descriptor
               lds_addr,                             // LDS byte address
               (uint32_t)ga,                         // global_addr[31:0]
               (uint32_t)((ga >> 32) & 0x01FFFFFFu) | (2u << 30) }; // [56:32]|type=2
  i32x8 g1 = { (int)0x00010000u,                     // wg_mask=0, data_size=2B
               0,                                    // abar=0, tensor_dim0[15:0]=0
               (int)0x4000u,                         // tensor_dim0=1<<30 (hi), td1 lo=0
               (int)((tile_d0_elems << 16) | 0x4000u), // td1=1<<30 (hi), tile_dim0
               (int)(tile_d1_rows & 0xFFFFu),        // tile_dim1, tile_dim2=0
               (int)row_stride_elems,                // tensor_dim0_stride[31:0]
               0, 0 };                               // stride0 hi / stride1 = 0
  i32x4 z4 = { 0, 0, 0, 0 };
#if __clang_major__ >= 23
  i32x8 z8 = { 0, 0, 0, 0, 0, 0, 0, 0 };
  __builtin_amdgcn_tensor_load_to_lds(g0, g1, z4, z4, z8, 0);
#else
  __builtin_amdgcn_tensor_load_to_lds(g0, g1, z4, z4, 0);
#endif
}

// ---------------------------------------------------------------------------
// fp32 -> bf16 streaming conversion
// ---------------------------------------------------------------------------
__global__ void f32_to_bf16_kernel(const float* __restrict__ src,
                                   __bf16* __restrict__ dst, int n) {
  int i = blockIdx.x * blockDim.x + threadIdx.x;
  if (i < n) dst[i] = f2bf(src[i]);
}

// ---------------------------------------------------------------------------
// QKV projection + fused RoPE + head scatter.
// qkv[s][e] = sum_d x[s][d] * Wqkv[e][d].  WG = 8 waves = 128x64 tile.
// Weight tile (64x32 bf16, 4KB) staged to LDS by TDM, double-buffered and
// shared by all 8 waves; A fragments stream from global per wave.
// All loads hoisted ahead of the 4 WMMAs so they clause + overlap and the
// WMMAs issue back-to-back.
// RoPE pairs (2i,2i+1) live in adjacent lanes of the C fragment -> shfl_xor(1).
// V is written transposed: Vt[(b,h)][d][s].
// ---------------------------------------------------------------------------
__global__ __launch_bounds__(256)
void qkv_rope_kernel(const __bf16* __restrict__ X, const __bf16* __restrict__ Wq,
                     const int* __restrict__ tpos,
                     __bf16* __restrict__ Q, __bf16* __restrict__ K,
                     __bf16* __restrict__ Vt) {
  __shared__ __align__(64) __bf16 btile[2][64 * 32];   // ping-pong weight tile

  const int lane = threadIdx.x & 31, wave = threadIdx.x >> 5;
  const int n = lane & 15, hi = lane >> 4;
  const int b    = blockIdx.z;
  const int row0 = blockIdx.y * 128 + wave * 16;
  const int col0 = blockIdx.x * 64;
  const __bf16* Xb     = X + (size_t)b * S * D;
  const __bf16* Wtile0 = Wq + (size_t)col0 * D;

  if (wave == 0)
    tdm_load_2d_to_lds(Wtile0, (uint32_t)(uintptr_t)&btile[0][0], 32, 64, D);

  v8f acc[4] = {};
  int buf = 0;
  for (int kk = 0; kk < D; kk += 32) {
    if (wave == 0) __builtin_amdgcn_s_wait_tensorcnt((short)0);
    __syncthreads();                       // tile `buf` published to all waves
    if (wave == 0 && kk + 32 < D)          // overlap next DMA with compute
      tdm_load_2d_to_lds(Wtile0 + kk + 32,
                         (uint32_t)(uintptr_t)&btile[buf ^ 1][0], 32, 64, D);

    v16bf a  = load_a_frag(Xb, row0, D, kk, lane);
    v16bf b0 = *(const v16bf*)&btile[buf][(0 * 16 + n) * 32 + hi * 16];
    v16bf b1 = *(const v16bf*)&btile[buf][(1 * 16 + n) * 32 + hi * 16];
    v16bf b2 = *(const v16bf*)&btile[buf][(2 * 16 + n) * 32 + hi * 16];
    v16bf b3 = *(const v16bf*)&btile[buf][(3 * 16 + n) * 32 + hi * 16];
    acc[0] = WMMA_BF16(a, b0, acc[0]);
    acc[1] = WMMA_BF16(a, b1, acc[1]);
    acc[2] = WMMA_BF16(a, b2, acc[2]);
    acc[3] = WMMA_BF16(a, b3, acc[3]);
    buf ^= 1;
  }

#pragma unroll
  for (int t = 0; t < 4; ++t) {
    const int e    = col0 + t * 16 + n;   // output feature in [0,3072)
    const int part = e >> 10;             // 0=q 1=k 2=v (uniform across wave)
    const int rem  = e & 1023;
    const int hd   = rem >> 6;            // head
    const int d    = rem & 63;            // dim within head
#pragma unroll
    for (int r = 0; r < 8; ++r) {
      const int s = row0 + r + hi * 8;
      float val = acc[t][r];
      if (part < 2) {
        float partner = __shfl_xor(val, 1, 32);   // paired even/odd channel
        int   pos = tpos[s];
        int   i   = d >> 1;
        // inv_freq = 10000^(-i/32) = 2^(-i*log2(10000)/32)
        float ang = (float)pos * exp2f(-(float)i * 0.41524101186092029f);
        float sn = __sinf(ang), cs = __cosf(ang);
        float res = val * cs + ((d & 1) ? partner * sn : -partner * sn);
        size_t idx = (((size_t)(b * H + hd) * S) + s) * DK + d;
        if (part == 0) Q[idx] = f2bf(res); else K[idx] = f2bf(res);
      } else {
        Vt[(((size_t)(b * H + hd) * DK) + d) * S + s] = f2bf(val);
      }
    }
  }
}

// ---------------------------------------------------------------------------
// Flash-style causal attention. 4 waves/WG; each wave owns a 16-row Q tile.
// Per 32-key step: 4 WMMA for S = Q*K^T, online softmax, LDS transpose of P
// (C layout -> A layout), 4 WMMA for O += P*V. Output written bf16 as
// O[b][s][h*64+d] ready for the output projection.
// ---------------------------------------------------------------------------
__global__ __launch_bounds__(128)
void attn_kernel(const __bf16* __restrict__ Q, const __bf16* __restrict__ Kx,
                 const __bf16* __restrict__ Vt, __bf16* __restrict__ O) {
  __shared__ __align__(32) __bf16 plds[4][16][32];

  const int lane = threadIdx.x & 31, wave = threadIdx.x >> 5;
  const int b = blockIdx.z, hd = blockIdx.y;
  const int q0 = blockIdx.x * 64 + wave * 16;

  const __bf16* Qbh = Q  + (size_t)(b * H + hd) * S * DK;
  const __bf16* Kbh = Kx + (size_t)(b * H + hd) * S * DK;
  const __bf16* Vbh = Vt + (size_t)(b * H + hd) * DK * S;

  const v16bf aq0 = load_a_frag(Qbh, q0, DK, 0,  lane);
  const v16bf aq1 = load_a_frag(Qbh, q0, DK, 32, lane);

  v8f o0 = {}, o1 = {}, o2 = {}, o3 = {};
  float mrow[8], lrow[8];
#pragma unroll
  for (int r = 0; r < 8; ++r) { mrow[r] = -1e30f; lrow[r] = 0.f; }

  const int n = lane & 15, hi = lane >> 4;
  const float LOG2E = 1.44269504088896340736f;

  for (int kb = 0; kb <= q0 + 15; kb += 32) {      // causal: keys <= q0+15
    // Hoist all 4 K fragments, then issue the 4 S-WMMAs back-to-back.
    v16bf bk00 = load_b_frag(Kbh, kb,      DK, 0,  lane);
    v16bf bk01 = load_b_frag(Kbh, kb,      DK, 32, lane);
    v16bf bk10 = load_b_frag(Kbh, kb + 16, DK, 0,  lane);
    v16bf bk11 = load_b_frag(Kbh, kb + 16, DK, 32, lane);
    v8f s0 = {}, s1 = {};
    s0 = WMMA_BF16(aq0, bk00, s0);
    s0 = WMMA_BF16(aq1, bk01, s0);
    s1 = WMMA_BF16(aq0, bk10, s1);
    s1 = WMMA_BF16(aq1, bk11, s1);

    float p0[8], p1[8], alpha[8];
#pragma unroll
    for (int r = 0; r < 8; ++r) {
      const int m = q0 + r + hi * 8;               // query row of this element
      float v0 = (kb + n      <= m) ? s0[r] * 0.125f : -1e30f;
      float v1 = (kb + 16 + n <= m) ? s1[r] * 0.125f : -1e30f;
      float rmax = fmaxf(v0, v1);
#pragma unroll
      for (int off = 1; off < 16; off <<= 1)       // stays inside 16-lane half
        rmax = fmaxf(rmax, __shfl_xor(rmax, off, 32));
      float mnew = fmaxf(mrow[r], rmax);
      alpha[r] = exp2f((mrow[r] - mnew) * LOG2E);
      p0[r]    = exp2f((v0 - mnew) * LOG2E);
      p1[r]    = exp2f((v1 - mnew) * LOG2E);
      float rs = p0[r] + p1[r];
#pragma unroll
      for (int off = 1; off < 16; off <<= 1)
        rs += __shfl_xor(rs, off, 32);
      lrow[r] = lrow[r] * alpha[r] + rs;
      mrow[r] = mnew;
    }
#pragma unroll
    for (int r = 0; r < 8; ++r) {
      o0[r] *= alpha[r]; o1[r] *= alpha[r]; o2[r] *= alpha[r]; o3[r] *= alpha[r];
    }

    // Transpose P through wave-private LDS slice: C layout -> A layout.
#pragma unroll
    for (int r = 0; r < 8; ++r) {
      plds[wave][r + hi * 8][n]      = f2bf(p0[r]);
      plds[wave][r + hi * 8][16 + n] = f2bf(p1[r]);
    }
    asm volatile("s_wait_dscnt 0x0" ::: "memory");  // wave-local LDS RAW fence
    AFrag pf;
    pf.h[0] = *(const v8bf*)&plds[wave][n][hi * 16];
    pf.h[1] = *(const v8bf*)&plds[wave][n][hi * 16 + 8];

    // Hoist all 4 V fragments, then the 4 O-WMMAs back-to-back.
    v16bf bv0 = load_b_frag(Vbh, 0,  S, kb, lane);
    v16bf bv1 = load_b_frag(Vbh, 16, S, kb, lane);
    v16bf bv2 = load_b_frag(Vbh, 32, S, kb, lane);
    v16bf bv3 = load_b_frag(Vbh, 48, S, kb, lane);
    o0 = WMMA_BF16(pf.v, bv0, o0);
    o1 = WMMA_BF16(pf.v, bv1, o1);
    o2 = WMMA_BF16(pf.v, bv2, o2);
    o3 = WMMA_BF16(pf.v, bv3, o3);
  }

#pragma unroll
  for (int r = 0; r < 8; ++r) {
    const int s = q0 + r + hi * 8;
    const float inv = 1.0f / lrow[r];
    size_t base = ((size_t)b * S + s) * D + hd * DK;
    O[base +  0 + n] = f2bf(o0[r] * inv);
    O[base + 16 + n] = f2bf(o1[r] * inv);
    O[base + 32 + n] = f2bf(o2[r] * inv);
    O[base + 48 + n] = f2bf(o3[r] * inv);
  }
}

// ---------------------------------------------------------------------------
// Output projection: out[b][s][e] = sum_f o[b][s][f] * Wo[e][f]  (f32 out)
// Same TDM-staged weight tile scheme as the QKV projection.
// ---------------------------------------------------------------------------
__global__ __launch_bounds__(256)
void out_gemm_kernel(const __bf16* __restrict__ A, const __bf16* __restrict__ W,
                     float* __restrict__ C, int M, int N, int Kd) {
  __shared__ __align__(64) __bf16 btile[2][64 * 32];

  const int lane = threadIdx.x & 31, wave = threadIdx.x >> 5;
  const int n = lane & 15, hi = lane >> 4;
  const int bz   = blockIdx.z;
  const int row0 = blockIdx.y * 128 + wave * 16;
  const int col0 = blockIdx.x * 64;
  const __bf16* Ab     = A + (size_t)bz * M * Kd;
  const __bf16* Wtile0 = W + (size_t)col0 * Kd;
  float*        Cb     = C + (size_t)bz * M * N;

  if (wave == 0)
    tdm_load_2d_to_lds(Wtile0, (uint32_t)(uintptr_t)&btile[0][0], 32, 64, Kd);

  v8f acc[4] = {};
  int buf = 0;
  for (int kk = 0; kk < Kd; kk += 32) {
    if (wave == 0) __builtin_amdgcn_s_wait_tensorcnt((short)0);
    __syncthreads();
    if (wave == 0 && kk + 32 < Kd)
      tdm_load_2d_to_lds(Wtile0 + kk + 32,
                         (uint32_t)(uintptr_t)&btile[buf ^ 1][0], 32, 64, Kd);

    v16bf a  = load_a_frag(Ab, row0, Kd, kk, lane);
    v16bf b0 = *(const v16bf*)&btile[buf][(0 * 16 + n) * 32 + hi * 16];
    v16bf b1 = *(const v16bf*)&btile[buf][(1 * 16 + n) * 32 + hi * 16];
    v16bf b2 = *(const v16bf*)&btile[buf][(2 * 16 + n) * 32 + hi * 16];
    v16bf b3 = *(const v16bf*)&btile[buf][(3 * 16 + n) * 32 + hi * 16];
    acc[0] = WMMA_BF16(a, b0, acc[0]);
    acc[1] = WMMA_BF16(a, b1, acc[1]);
    acc[2] = WMMA_BF16(a, b2, acc[2]);
    acc[3] = WMMA_BF16(a, b3, acc[3]);
    buf ^= 1;
  }

#pragma unroll
  for (int t = 0; t < 4; ++t)
#pragma unroll
    for (int r = 0; r < 8; ++r)
      Cb[(size_t)(row0 + r + hi * 8) * N + col0 + t * 16 + n] = acc[t][r];
}

// ---------------------------------------------------------------------------
// Launch
// ---------------------------------------------------------------------------
extern "C" void kernel_launch(void* const* d_in, const int* in_sizes, int n_in,
                              void* d_out, int out_size, void* d_ws, size_t ws_size,
                              hipStream_t stream) {
  (void)in_sizes; (void)n_in; (void)out_size; (void)ws_size;
  const float* x    = (const float*)d_in[0];
  const float* Wqkv = (const float*)d_in[1];
  const float* Wo   = (const float*)d_in[2];
  const int*   tpos = (const int*)d_in[3];
  float*       out  = (float*)d_out;

  char* ws = (char*)d_ws;
  size_t off = 0;
  auto alloc = [&](size_t bytes) -> void* {
    void* p = ws + off;
    off += (bytes + 255) & ~(size_t)255;
    return p;
  };
  __bf16* Xb  = (__bf16*)alloc((size_t)Bb * S * D * 2);      // 16 MB
  __bf16* Wqb = (__bf16*)alloc((size_t)N3 * D * 2);          //  6 MB
  __bf16* Wob = (__bf16*)alloc((size_t)D * D * 2);           //  2 MB
  __bf16* Qb  = (__bf16*)alloc((size_t)Bb * H * S * DK * 2); // 16 MB
  __bf16* Kb  = (__bf16*)alloc((size_t)Bb * H * S * DK * 2); // 16 MB
  __bf16* Vtb = (__bf16*)alloc((size_t)Bb * H * DK * S * 2); // 16 MB
  __bf16* Ob  = (__bf16*)alloc((size_t)Bb * S * D * 2);      // 16 MB

  const int nx = Bb * S * D;
  f32_to_bf16_kernel<<<(nx + 255) / 256, 256, 0, stream>>>(x, Xb, nx);
  const int nw = N3 * D;
  f32_to_bf16_kernel<<<(nw + 255) / 256, 256, 0, stream>>>(Wqkv, Wqb, nw);
  const int no = D * D;
  f32_to_bf16_kernel<<<(no + 255) / 256, 256, 0, stream>>>(Wo, Wob, no);

  qkv_rope_kernel<<<dim3(N3 / 64, S / 128, Bb), 256, 0, stream>>>(Xb, Wqb, tpos,
                                                                  Qb, Kb, Vtb);
  attn_kernel<<<dim3(S / 64, H, Bb), 128, 0, stream>>>(Qb, Kb, Vtb, Ob);
  out_gemm_kernel<<<dim3(D / 64, S / 128, Bb), 256, 0, stream>>>(Ob, Wob, out,
                                                                 S, D, D);
}
